// BernoulliMixture_30562987278389
// MI455X (gfx1250) — matrix operand order repaired and checked by reading, not compiled, not run on hardware
//
#include <hip/hip_runtime.h>
#include <hip/hip_bf16.h>

// ---------------------------------------------------------------------------
// BernoulliMixture on MI455X (gfx1250), wave32 + WMMA f16 + async LDS loads.
// B=4, L=1024 (M=4096 rows), IN=512, HID=1024, K=32, O=784.
// ---------------------------------------------------------------------------

#define M_ROWS 4096
#define IN_DIM 512
#define HID    1024
#define KCOMP  32
#define ODIM   784

typedef __attribute__((ext_vector_type(16))) _Float16 v16h;
typedef __attribute__((ext_vector_type(8)))  _Float16 v8h;
typedef __attribute__((ext_vector_type(4)))  _Float16 v4h;
typedef __attribute__((ext_vector_type(8)))  float    v8f;

// A-fragment (16-bit A 16x32): this lane needs halves [p,p+8) and [p+16,p+24)
// where p already includes the lane's klo offset (0 or 8).
static __device__ inline v16h load_frag_a(const _Float16* p) {
    v16h a;
    ((v8h*)&a)[0] = *(const v8h*)(p);
    ((v8h*)&a)[1] = *(const v8h*)(p + 16);
    return a;
}
// B-fragment (16-bit B 32x16): this lane needs halves [p,p+8) and [p+16,p+24)
// where p already includes the lane's koff offset (0 or 8).
static __device__ inline v16h load_frag_b(const _Float16* p) {
    v16h b;
    ((v8h*)&b)[0] = *(const v8h*)(p);
    ((v8h*)&b)[1] = *(const v8h*)(p + 16);
    return b;
}

// CDNA5 async memory->LDS copy (VGLOBAL, tracked by ASYNCcnt).
// lds_off is a byte offset relative to the workgroup LDS base (no static LDS
// in the kernel, so dynamic shared memory starts at offset 0).
static __device__ inline void async_g2l_b128(unsigned int lds_off, const void* gaddr) {
    asm volatile("global_load_async_to_lds_b128 %0, %1, off"
                 :: "v"(lds_off), "v"((unsigned long long)(uintptr_t)gaddr)
                 : "memory");
}
static __device__ inline void wait_asynccnt0() {
    asm volatile("s_wait_asynccnt 0" ::: "memory");
}

// ---------------------------------------------------------------------------
// 0) f32 -> f16 convert, 4 elements/thread
// ---------------------------------------------------------------------------
__global__ void cvt_f32_f16(const float* __restrict__ src,
                            _Float16* __restrict__ dst, int n4) {
    int i = blockIdx.x * blockDim.x + threadIdx.x;
    if (i < n4) {
        float4 v = ((const float4*)src)[i];
        v4h h;
        h.x = (_Float16)v.x; h.y = (_Float16)v.y;
        h.z = (_Float16)v.z; h.w = (_Float16)v.w;
        ((v4h*)dst)[i] = h;
    }
}

// ---------------------------------------------------------------------------
// 1) h = leaky_relu(X @ W1^T + b1)  : one wave computes one 16x16 tile.
//    grid = (4096/16, 1024/16), block = 32 (one wave).
// ---------------------------------------------------------------------------
__global__ __launch_bounds__(32) void gemm1_h(const _Float16* __restrict__ X,
                                              const _Float16* __restrict__ W1,
                                              const float* __restrict__ b1,
                                              _Float16* __restrict__ H) {
    const int lane = threadIdx.x;
    const int lm   = lane & 15;
    const int hi   = lane >> 4;           // 0 or 1
    const int m0   = blockIdx.x * 16;
    const int n0   = blockIdx.y * 16;

    const _Float16* arow = X  + (size_t)(m0 + lm) * IN_DIM + (hi ? 8 : 0);
    const _Float16* brow = W1 + (size_t)(n0 + lm) * IN_DIM + (hi ? 8 : 0);

    v8f acc = {};
#pragma unroll 4
    for (int kk = 0; kk < IN_DIM; kk += 32) {
        v16h a = load_frag_a(arow + kk);
        v16h b = load_frag_b(brow + kk);
        acc = __builtin_amdgcn_wmma_f32_16x16x32_f16(
            false, a, false, b, (short)0, acc, false, false);
    }

    const int n  = n0 + lm;
    const float bv = b1[n];
    const int mb = m0 + hi * 8;
#pragma unroll
    for (int r = 0; r < 8; ++r) {
        float v = acc[r] + bv;
        v = (v >= 0.0f) ? v : 0.01f * v;
        H[(size_t)(mb + r) * HID + n] = (_Float16)v;
    }
}

// ---------------------------------------------------------------------------
// 2) z = softmax(h @ Wz^T + bz) over K=32. One block (256 thr) per row.
// ---------------------------------------------------------------------------
__global__ __launch_bounds__(256) void gate_softmax(const _Float16* __restrict__ H,
                                                    const _Float16* __restrict__ Wz,
                                                    const float* __restrict__ bz,
                                                    float* __restrict__ Z) {
    __shared__ float logits[KCOMP];
    const int row = blockIdx.x;
    const int tid = threadIdx.x;
    const int k   = tid >> 3;   // 0..31
    const int s   = tid & 7;    // 0..7

    const _Float16* h = H  + (size_t)row * HID;
    const _Float16* w = Wz + (size_t)k * HID;
    float sum = 0.0f;
    for (int i = s; i < HID; i += 8)
        sum += (float)h[i] * (float)w[i];
    // reduce across the 8 lanes of this k-group (contiguous lanes in a wave)
    sum += __shfl_down(sum, 4, 8);
    sum += __shfl_down(sum, 2, 8);
    sum += __shfl_down(sum, 1, 8);
    if (s == 0) logits[k] = sum + bz[k];
    __syncthreads();

    if (tid < KCOMP) {
        float v  = logits[tid];
        float mx = v;
#pragma unroll
        for (int off = 16; off > 0; off >>= 1)
            mx = fmaxf(mx, __shfl_xor(mx, off, 32));
        float e = __expf(v - mx);
        float sm = e;
#pragma unroll
        for (int off = 16; off > 0; off >>= 1)
            sm += __shfl_xor(sm, off, 32);
        Z[(size_t)row * KCOMP + tid] = e / sm;
    }
}

// ---------------------------------------------------------------------------
// 3) Fused d-GEMM + sigmoid + mixture:
//    out[m, o] = sum_k z[m,k] * sigmoid(h[m,:]·Wd[k*784+o,:] + bd[k*784+o])
//    grid = (4096/128, 784/16), block = 256 (8 waves).
//    LDS: 128x1024 f16 h-tile (256KB, CDNA5 320KB LDS) + 128x32 f32 z-tile,
//    staged with GLOBAL_LOAD_ASYNC_TO_LDS_B128 (ASYNCcnt path).
// ---------------------------------------------------------------------------
#define MTILE 128
#define HS_BYTES (MTILE * HID * 2)
#define ZS_BYTES (MTILE * KCOMP * 4)
#define DMIX_LDS_BYTES (HS_BYTES + ZS_BYTES)

__global__ __launch_bounds__(256) void dmix(const _Float16* __restrict__ H,
                                            const _Float16* __restrict__ Wd,
                                            const float* __restrict__ bd,
                                            const float* __restrict__ Z,
                                            float* __restrict__ OUT) {
    extern __shared__ __align__(16) char smem[];
    _Float16* hs = (_Float16*)smem;                 // 128*1024 halves, LDS off 0
    float*    zs = (float*)(smem + HS_BYTES);       // 128*32 floats

    const int tid  = threadIdx.x;
    const int row0 = blockIdx.x * MTILE;
    const int o0   = blockIdx.y * 16;

    // Async-stage h tile (256KB) and z tile (16KB): memory -> LDS directly,
    // 16B per lane per issue, no VGPR round trip (tracked by ASYNCcnt).
    {
        const char* hsrc = (const char*)(H + (size_t)row0 * HID);
        for (int i = tid; i < HS_BYTES / 16; i += 256)          // 16384 vecs
            async_g2l_b128((unsigned int)(i * 16), hsrc + (size_t)i * 16);
        const char* zsrc = (const char*)(Z + (size_t)row0 * KCOMP);
        for (int i = tid; i < ZS_BYTES / 16; i += 256)          // 1024 vecs
            async_g2l_b128((unsigned int)(HS_BYTES + i * 16), zsrc + (size_t)i * 16);
        wait_asynccnt0();
    }
    __syncthreads();

    const int wave = tid >> 5;
    const int lane = tid & 31;
    const int lm   = lane & 15;
    const int hi   = lane >> 4;
    const int msub = wave * 16;
    const int n    = o0 + lm;                       // o column

    const _Float16* arow = hs + (size_t)(msub + lm) * HID + (hi ? 8 : 0);

    v8f oacc = {};
    for (int k = 0; k < KCOMP; ++k) {
        const _Float16* brow = Wd + (size_t)(k * ODIM + n) * HID + (hi ? 8 : 0);
        if (k + 1 < KCOMP)   // warm L2->L0 path for next component's row
            __builtin_prefetch(Wd + (size_t)((k + 1) * ODIM + n) * HID, 0, 0);

        v8f acc = {};
#pragma unroll 4
        for (int kk = 0; kk < HID; kk += 32) {
            v16h a = load_frag_a(arow + kk);   // ds_load_b128 x2
            v16h b = load_frag_b(brow + kk);   // global_load_b128 x2
            acc = __builtin_amdgcn_wmma_f32_16x16x32_f16(
                false, a, false, b, (short)0, acc, false, false);
        }

        const float bv = bd[k * ODIM + n];
#pragma unroll
        for (int r = 0; r < 8; ++r) {
            const int m  = msub + r + hi * 8;  // row within tile
            const float zv = zs[m * KCOMP + k];
            const float x  = acc[r] + bv;
            const float dg = 1.0f / (1.0f + __expf(-x));
            oacc[r] += zv * dg;
        }
    }

#pragma unroll
    for (int r = 0; r < 8; ++r) {
        const int m = row0 + msub + r + hi * 8;
        OUT[(size_t)m * ODIM + n] = oacc[r];
    }
}

// ---------------------------------------------------------------------------
// Host launcher
// ---------------------------------------------------------------------------
extern "C" void kernel_launch(void* const* d_in, const int* in_sizes, int n_in,
                              void* d_out, int out_size, void* d_ws, size_t ws_size,
                              hipStream_t stream) {
    const float* input = (const float*)d_in[0];   // [4,1024,512]
    const float* w_i2h = (const float*)d_in[1];   // [1024,512]
    const float* b_i2h = (const float*)d_in[2];   // [1024]
    const float* w_h2z = (const float*)d_in[3];   // [32,1024]
    const float* b_h2z = (const float*)d_in[4];   // [32]
    const float* w_h2d = (const float*)d_in[5];   // [25088,1024]
    const float* b_h2d = (const float*)d_in[6];   // [25088]
    float* out = (float*)d_out;                   // [4,1024,784]

    char* ws = (char*)d_ws;
    _Float16* wd16 = (_Float16*)ws; ws += (size_t)KCOMP * ODIM * HID * 2; // 51.4MB
    _Float16* h16  = (_Float16*)ws; ws += (size_t)M_ROWS * HID * 2;       // 8MB
    _Float16* x16  = (_Float16*)ws; ws += (size_t)M_ROWS * IN_DIM * 2;    // 4MB
    _Float16* w116 = (_Float16*)ws; ws += (size_t)HID * IN_DIM * 2;       // 1MB
    _Float16* wz16 = (_Float16*)ws; ws += (size_t)KCOMP * HID * 2;        // 64KB
    float*    zbuf = (float*)ws;    ws += (size_t)M_ROWS * KCOMP * 4;     // 512KB

    auto cvt = [&](const float* s, _Float16* d, size_t n) {
        int n4 = (int)(n / 4);
        cvt_f32_f16<<<(n4 + 255) / 256, 256, 0, stream>>>(s, d, n4);
    };
    cvt(input, x16, (size_t)M_ROWS * IN_DIM);
    cvt(w_i2h, w116, (size_t)HID * IN_DIM);
    cvt(w_h2z, wz16, (size_t)KCOMP * HID);
    cvt(w_h2d, wd16, (size_t)KCOMP * ODIM * HID);

    gemm1_h<<<dim3(M_ROWS / 16, HID / 16), 32, 0, stream>>>(x16, w116, b_i2h, h16);
    gate_softmax<<<M_ROWS, 256, 0, stream>>>(h16, wz16, b_h2z, zbuf);

    (void)hipFuncSetAttribute((const void*)dmix,
                              hipFuncAttributeMaxDynamicSharedMemorySize,
                              (int)DMIX_LDS_BYTES);
    dmix<<<dim3(M_ROWS / MTILE, ODIM / 16), 256, DMIX_LDS_BYTES, stream>>>(
        h16, wd16, b_h2d, zbuf, out);
}
